// GAT_interpolation_48198122995731
// MI455X (gfx1250) — compile-verified, matrix operand order checked by reading, not run on hardware
//
#include <hip/hip_runtime.h>
#include <hip/hip_bf16.h>

// ---------------------------------------------------------------------------
// GAT (2-layer) + final linear for MI455X (gfx1250, wave32).
// GEMMs use v_wmma_f32_16x16x32_bf16 with pre-swizzled B fragments (no LDS,
// no per-step swizzle work); edge softmax/aggregation uses L2-resident
// atomics (all aggregation state fits in the 192MB L2).
// ---------------------------------------------------------------------------

#define NEG_SLOPE 0.2f

typedef __attribute__((ext_vector_type(16))) __bf16 v16bf;
typedef __attribute__((ext_vector_type(8)))  float  v8f;

// ---------------- order-preserving float<->uint encoding for atomicMax ------
__device__ __forceinline__ unsigned f2ord(float f) {
    unsigned u = __float_as_uint(f);
    return (u & 0x80000000u) ? ~u : (u | 0x80000000u);
}
__device__ __forceinline__ float ord2f(unsigned e) {
    unsigned u = (e & 0x80000000u) ? (e ^ 0x80000000u) : ~e;
    return __uint_as_float(u);
}

// ---------------------------------------------------------------------------
// Pack W[K,32] (f32, row-major) into WMMA B fragments (bf16), lane-contiguous:
//   frag f = (kstep, coltile):  out[f*512 + lane*16 + j] = W[k(j,lane)][col]
// so the GEMM can load each fragment as one 32-byte v16bf per lane
// (two coalesced global_load_b128).
// K striping mirrors the 16-bit A-matrix layout (ISA 7.12.2):
//   j in 0..7  -> k = kstep*32 + half*8 + j
//   j in 8..15 -> k = kstep*32 + 16 + half*8 + (j-8)
// ---------------------------------------------------------------------------
__global__ __launch_bounds__(256) void gat_pack_bfrag(
        const float* __restrict__ W, __bf16* __restrict__ out, int K) {
    const int total = (K / 32) * 2 * 32 * 16;      // frags * lanes * elems
    const int idx = blockIdx.x * blockDim.x + threadIdx.x;
    if (idx >= total) return;
    const int j     = idx & 15;
    const int lane  = (idx >> 4) & 31;
    const int f     = idx >> 9;
    const int kstep = f >> 1;
    const int ct    = f & 1;
    const int half  = lane >> 4;
    const int col   = ct * 16 + (lane & 15);
    const int k     = kstep * 32 + ((j < 8) ? (half * 8 + j)
                                            : (16 + half * 8 + (j - 8)));
    out[idx] = (__bf16)W[k * 32 + col];
}

// ---------------------------------------------------------------------------
// GEMM: Out[N,32] = X[N,K] @ W[K,32] (+bias), B pre-packed as bf16 fragments.
// Block = 256 threads = 8 waves; each wave computes a 16x32 output tile via
// two v_wmma_f32_16x16x32_bf16 accumulators per K-step. K templated so the
// K-loop fully unrolls.
// ---------------------------------------------------------------------------
template <int K>
__global__ __launch_bounds__(256) void gat_gemm32_wmma(
        const float* __restrict__ X,
        const __bf16* __restrict__ Bfrag,
        const float* __restrict__ bias,
        float* __restrict__ Out,
        int N) {
    const int waveId = threadIdx.x >> 5;
    const int lane   = threadIdx.x & 31;
    const int row0   = blockIdx.x * 128 + waveId * 16;
    if (row0 + 16 > N) return;                        // wave-uniform: EXEC stays all-1s

    const int mrow = lane & 15;                       // A row within tile
    const int half = lane >> 4;                       // K half selector
    const int ncol = lane & 15;                       // B/C column within tile

    // Preload bias so the writeback loop is branch-free straight-line stores.
    float bv0 = 0.0f, bv1 = 0.0f;
    if (bias) { bv0 = bias[ncol]; bv1 = bias[16 + ncol]; }

    v8f acc0 = {};
    v8f acc1 = {};
    const float* xrow = X + (size_t)(row0 + mrow) * K;
    const v16bf* bp   = (const v16bf*)Bfrag;          // 32B-aligned fragments

#pragma unroll
    for (int ks = 0; ks < K / 32; ++ks) {
        // A fragment: 16-bit A 16x32 layout (ISA 7.12.2):
        // lane half h: elems 0..7 -> K = 32ks + 8h + j, elems 8..15 -> +16
        v16bf a;
        const float* xp = xrow + ks * 32 + half * 8;
#pragma unroll
        for (int j = 0; j < 8; ++j) {
            a[j]     = (__bf16)xp[j];
            a[8 + j] = (__bf16)xp[16 + j];
        }
        const v16bf b0 = bp[(ks * 2 + 0) * 32 + lane];
        const v16bf b1 = bp[(ks * 2 + 1) * 32 + lane];
        acc0 = __builtin_amdgcn_wmma_f32_16x16x32_bf16(false, a, false, b0,
                                                       (short)0, acc0, false, false);
        acc1 = __builtin_amdgcn_wmma_f32_16x16x32_bf16(false, a, false, b1,
                                                       (short)0, acc1, false, false);
    }

    // C/D layout: VGPR r, lanes 0-15 -> M=r, lanes 16-31 -> M=r+8, N=lane%16
#pragma unroll
    for (int r = 0; r < 8; ++r) {
        const int node = row0 + r + 8 * half;
        Out[(size_t)node * 32 + ncol]      = acc0[r] + bv0;
        Out[(size_t)node * 32 + 16 + ncol] = acc1[r] + bv1;
    }
}

// ---------------------------------------------------------------------------
// Per-node attention scores: as_[n] = h[n,:]·a_src ; ad_[n] = h[n,:]·a_dst
// One wave32 per node, lane = channel, shuffle reduction.
// ---------------------------------------------------------------------------
__global__ __launch_bounds__(256) void gat_scores(
        const float* __restrict__ h,
        const float* __restrict__ a_s,
        const float* __restrict__ a_d,
        float* __restrict__ as_, float* __restrict__ ad_,
        int N) {
    const int node = (blockIdx.x * blockDim.x + threadIdx.x) >> 5;
    const int lane = threadIdx.x & 31;
    if (node >= N) return;
    const float hv = h[(size_t)node * 32 + lane];
    float ps = hv * a_s[lane];
    float pd = hv * a_d[lane];
#pragma unroll
    for (int off = 16; off > 0; off >>= 1) {
        ps += __shfl_down(ps, off, 32);
        pd += __shfl_down(pd, off, 32);
    }
    if (lane == 0) { as_[node] = ps; ad_[node] = pd; }
}

// ---------------------------------------------------------------------------
// Edge pass 1: segment max of leaky_relu(as_[src]+ad_[dst]) over dst.
// ---------------------------------------------------------------------------
__global__ __launch_bounds__(256) void gat_edge_max(
        const int* __restrict__ ei, int E, int N,
        const float* __restrict__ as_,
        const float* __restrict__ ad_,
        unsigned* __restrict__ emax) {
    const int e = blockIdx.x * blockDim.x + threadIdx.x;
    const int Et = E + N;
    if (e >= Et) return;
    int s, d;
    if (e < E) { s = ei[e]; d = ei[E + e]; } else { s = d = e - E; }
    float v = as_[s] + ad_[d];
    v = (v > 0.0f) ? v : NEG_SLOPE * v;
    atomicMax(&emax[d], f2ord(v));
}

// ---------------------------------------------------------------------------
// Edge pass 2: e_exp = exp(e - max[dst]); segment sum over dst.
// ---------------------------------------------------------------------------
__global__ __launch_bounds__(256) void gat_edge_exp(
        const int* __restrict__ ei, int E, int N,
        const float* __restrict__ as_,
        const float* __restrict__ ad_,
        const unsigned* __restrict__ emax,
        float* __restrict__ eexp,
        float* __restrict__ esum) {
    const int e = blockIdx.x * blockDim.x + threadIdx.x;
    const int Et = E + N;
    if (e >= Et) return;
    int s, d;
    if (e < E) { s = ei[e]; d = ei[E + e]; } else { s = d = e - E; }
    float v = as_[s] + ad_[d];
    v = (v > 0.0f) ? v : NEG_SLOPE * v;
    const float ex = __expf(v - ord2f(emax[d]));
    eexp[e] = ex;
    atomicAdd(&esum[d], ex);
}

// ---------------------------------------------------------------------------
// Edge pass 3: agg[dst,:] += alpha * h[src,:]. One wave32 per edge,
// lane = channel -> coalesced 128B row read + one f32 atomic per lane.
// ---------------------------------------------------------------------------
__global__ __launch_bounds__(256) void gat_edge_agg(
        const int* __restrict__ ei, int E, int N,
        const float* __restrict__ eexp,
        const float* __restrict__ esum,
        const float* __restrict__ h,
        float* __restrict__ agg) {
    const long long wave = (long long)(blockIdx.x) * (blockDim.x >> 5)
                         + (threadIdx.x >> 5);
    const int lane = threadIdx.x & 31;
    const int Et = E + N;
    if (wave >= Et) return;
    const int e = (int)wave;
    int s, d;
    if (e < E) { s = ei[e]; d = ei[E + e]; } else { s = d = e - E; }
    const float alpha = eexp[e] / (esum[d] + 1e-16f);
    atomicAdd(&agg[(size_t)d * 32 + lane], alpha * h[(size_t)s * 32 + lane]);
}

// ---------------------------------------------------------------------------
// In-place bias + ReLU: x = max(x + b[c], 0)
// ---------------------------------------------------------------------------
__global__ __launch_bounds__(256) void gat_bias_relu(
        float* __restrict__ x, const float* __restrict__ b, int total) {
    const int i = blockIdx.x * blockDim.x + threadIdx.x;
    if (i >= total) return;
    const float v = x[i] + b[i & 31];
    x[i] = v > 0.0f ? v : 0.0f;
}

// ---------------------------------------------------------------------------
extern "C" void kernel_launch(void* const* d_in, const int* in_sizes, int n_in,
                              void* d_out, int out_size, void* d_ws, size_t ws_size,
                              hipStream_t stream) {
    const float* x        = (const float*)d_in[0];
    const int*   ei       = (const int*)  d_in[1];
    const float* W1       = (const float*)d_in[2];
    const float* a_src1   = (const float*)d_in[3];
    const float* a_dst1   = (const float*)d_in[4];
    const float* b1       = (const float*)d_in[5];
    const float* W2       = (const float*)d_in[6];
    const float* a_src2   = (const float*)d_in[7];
    const float* a_dst2   = (const float*)d_in[8];
    const float* b2       = (const float*)d_in[9];
    const float* Wf       = (const float*)d_in[10];
    const float* bf       = (const float*)d_in[11];

    const int N  = in_sizes[0] / 128;   // 100000
    const int E  = in_sizes[1] / 2;     // 3200000
    const int Et = E + N;               // edges incl. self-loops

    // Workspace layout (floats)
    float*    h    = (float*)d_ws;                 // N*32
    float*    agg  = h    + (size_t)N * 32;        // N*32 (becomes next x)
    float*    as_  = agg  + (size_t)N * 32;        // N
    float*    ad_  = as_  + N;                     // N
    float*    esum = ad_  + N;                     // N
    unsigned* emax = (unsigned*)(esum + N);        // N
    float*    eexp = (float*)(emax + N);           // Et
    // bf16 B-fragment store, 32B aligned for v16bf loads
    size_t fragOff = (((size_t)N * 68 + Et) + 7) & ~(size_t)7;  // in floats
    __bf16* fragW1 = (__bf16*)((float*)d_ws + fragOff);         // 4096 bf16
    __bf16* fragW2 = fragW1 + 4096;                             // 1024 bf16
    __bf16* fragWf = fragW2 + 1024;                             // 1024 bf16

    const int TB        = 256;
    const int gemmGrid  = (N + 127) / 128;
    const int edgeGrid  = (Et + TB - 1) / TB;
    const int waveGridE = (int)(((long long)Et * 32 + TB - 1) / TB);
    const int waveGridN = (int)(((long long)N  * 32 + TB - 1) / TB);
    const int nbGrid    = (N * 32 + TB - 1) / TB;

    // Pre-swizzle all three weight matrices into WMMA B fragments.
    gat_pack_bfrag<<<(4096 + TB - 1) / TB, TB, 0, stream>>>(W1, fragW1, 128);
    gat_pack_bfrag<<<(1024 + TB - 1) / TB, TB, 0, stream>>>(W2, fragW2, 32);
    gat_pack_bfrag<<<(1024 + TB - 1) / TB, TB, 0, stream>>>(Wf, fragWf, 32);

    // ---------------- Layer 1 (Fin = 128) ----------------
    gat_gemm32_wmma<128><<<gemmGrid, TB, 0, stream>>>(x, fragW1, nullptr, h, N);
    gat_scores<<<waveGridN, TB, 0, stream>>>(h, a_src1, a_dst1, as_, ad_, N);
    hipMemsetAsync(emax, 0, (size_t)N * 4, stream);   // ord-encoding minimum
    hipMemsetAsync(esum, 0, (size_t)N * 4, stream);
    hipMemsetAsync(agg,  0, (size_t)N * 32 * 4, stream);
    gat_edge_max<<<edgeGrid, TB, 0, stream>>>(ei, E, N, as_, ad_, emax);
    gat_edge_exp<<<edgeGrid, TB, 0, stream>>>(ei, E, N, as_, ad_, emax, eexp, esum);
    gat_edge_agg<<<waveGridE, TB, 0, stream>>>(ei, E, N, eexp, esum, h, agg);
    gat_bias_relu<<<nbGrid, TB, 0, stream>>>(agg, b1, N * 32);   // agg = x2

    // ---------------- Layer 2 (Fin = 32) ----------------
    gat_gemm32_wmma<32><<<gemmGrid, TB, 0, stream>>>(agg, fragW2, nullptr, h, N);
    gat_scores<<<waveGridN, TB, 0, stream>>>(h, a_src2, a_dst2, as_, ad_, N);
    hipMemsetAsync(emax, 0, (size_t)N * 4, stream);
    hipMemsetAsync(esum, 0, (size_t)N * 4, stream);
    hipMemsetAsync(agg,  0, (size_t)N * 32 * 4, stream);
    gat_edge_max<<<edgeGrid, TB, 0, stream>>>(ei, E, N, as_, ad_, emax);
    gat_edge_exp<<<edgeGrid, TB, 0, stream>>>(ei, E, N, as_, ad_, emax, eexp, esum);
    gat_edge_agg<<<waveGridE, TB, 0, stream>>>(ei, E, N, eexp, esum, h, agg);
    gat_bias_relu<<<nbGrid, TB, 0, stream>>>(agg, b2, N * 32);   // agg = h2

    // ---------------- Final linear: out = h2 @ Wf + bf ----------------
    gat_gemm32_wmma<32><<<gemmGrid, TB, 0, stream>>>(agg, fragWf, bf, (float*)d_out, N);
}
